// GPT_43181601194554
// MI455X (gfx1250) — matrix operand order repaired and checked by reading, not compiled, not run on hardware
//
#include <hip/hip_runtime.h>
#include <hip/hip_bf16.h>

typedef __attribute__((ext_vector_type(16))) _Float16 v16h;
typedef __attribute__((ext_vector_type(8)))  _Float16 v8h;
typedef __attribute__((ext_vector_type(8)))  float    v8f;
typedef __attribute__((ext_vector_type(4)))  float    v4f;

#define CAT16(a, b) __builtin_shufflevector((a), (b), 0,1,2,3,4,5,6,7,8,9,10,11,12,13,14,15)
#define WMMA_F16(a, b, c) \
  __builtin_amdgcn_wmma_f32_16x16x32_f16(false, (a), false, (b), (short)0, (c), false, false)

// ---------------------------------------------------------------------------
// Problem constants
// ---------------------------------------------------------------------------
#define BB   2
#define TT   2048
#define CC   1024
#define NH_  16
#define NKV_ 4
#define HD_  64
#define QKVN 1536          // C + 2*NKV*HD

// ---------------------------------------------------------------------------
// f16-WMMA GEMM:  C[M,N] = A[M,K] * B[K,N]   (fp32 in/out, f16 compute)
// N,K compile-time -> immediate-offset addressing in loads/stores.
// Block tile 128x128, 8 waves (4 M x 2 N), wave tile 32x64, k-step 32.
// Software-pipelined: prefetch next k-tile into VGPRs while WMMA runs.
// ---------------------------------------------------------------------------
template <int N, int K>
__global__ __launch_bounds__(256)
void gemm_wmma_f16(const float* __restrict__ A, const float* __restrict__ B,
                   float* __restrict__ C) {
  __shared__ _Float16 As[128][40];   // [m][k], row padded to 40 halfs (80B)
  __shared__ _Float16 Bst[128][40];  // transposed: [n][k]

  const int tid  = threadIdx.x;
  const int lane = tid & 31;
  const int wave = tid >> 5;
  const int m0   = blockIdx.y * 128;
  const int n0   = blockIdx.x * 128;
  const int wtm  = (wave >> 1) * 32;   // 0,32,64,96
  const int wtn  = (wave & 1) * 64;    // 0,64

  const int am   = lane & 15;          // row (A) / col (B) within 16
  const int koff = (lane >> 4) << 3;   // A-frag K chunk base: 0 or 8
  const int bk   = (lane >> 4) << 4;   // unused; B-frag also uses koff (0 or 8)

  // per-thread global->LDS slices (constant across k-steps)
  int rA[4], cA[4], rB[4], cB[4];
#pragma unroll
  for (int it = 0; it < 4; ++it) {
    int idx = tid + it * 256;          // 0..1023
    rA[it] = idx >> 3;                 // 0..127 (m)
    cA[it] = (idx & 7) << 2;           // 0..28  (k)
    rB[it] = idx >> 5;                 // 0..31  (k)
    cB[it] = (idx & 31) << 2;          // 0..124 (n)
  }

  v8f acc[2][4] = {};
  v4f ar[4], br[4];

  // prologue: prefetch k-tile 0 into registers
#pragma unroll
  for (int it = 0; it < 4; ++it) {
    ar[it] = *(const v4f*)(A + (size_t)(m0 + rA[it]) * K + cA[it]);
    br[it] = *(const v4f*)(B + (size_t)rB[it] * N + n0 + cB[it]);
  }

  for (int k0 = 0; k0 < K; k0 += 32) {
    // commit prefetched tile to LDS (f32 -> f16)
#pragma unroll
    for (int it = 0; it < 4; ++it) {
      As[rA[it]][cA[it] + 0] = (_Float16)ar[it][0];
      As[rA[it]][cA[it] + 1] = (_Float16)ar[it][1];
      As[rA[it]][cA[it] + 2] = (_Float16)ar[it][2];
      As[rA[it]][cA[it] + 3] = (_Float16)ar[it][3];
      Bst[cB[it] + 0][rB[it]] = (_Float16)br[it][0];
      Bst[cB[it] + 1][rB[it]] = (_Float16)br[it][1];
      Bst[cB[it] + 2][rB[it]] = (_Float16)br[it][2];
      Bst[cB[it] + 3][rB[it]] = (_Float16)br[it][3];
    }
    __syncthreads();

    // prefetch next k-tile while computing this one
    if (k0 + 32 < K) {
#pragma unroll
      for (int it = 0; it < 4; ++it) {
        ar[it] = *(const v4f*)(A + (size_t)(m0 + rA[it]) * K + (k0 + 32) + cA[it]);
        br[it] = *(const v4f*)(B + (size_t)(k0 + 32 + rB[it]) * N + n0 + cB[it]);
      }
    }

    v16h afr[2];
#pragma unroll
    for (int mi = 0; mi < 2; ++mi) {
      const _Float16* ap = &As[wtm + mi * 16 + am][0];
      v8h a0 = *(const v8h*)(ap + koff);
      v8h a1 = *(const v8h*)(ap + koff + 16);
      afr[mi] = CAT16(a0, a1);
    }
    v16h bfr[4];
#pragma unroll
    for (int ni = 0; ni < 4; ++ni) {
      const _Float16* bp = &Bst[wtn + ni * 16 + am][0];
      v8h b0 = *(const v8h*)(bp + koff);
      v8h b1 = *(const v8h*)(bp + koff + 16);
      bfr[ni] = CAT16(b0, b1);
    }
#pragma unroll
    for (int mi = 0; mi < 2; ++mi)
#pragma unroll
      for (int ni = 0; ni < 4; ++ni)
        acc[mi][ni] = WMMA_F16(afr[mi], bfr[ni], acc[mi][ni]);

    __syncthreads();
  }

  // Epilogue: D layout — lane = N col (mod 16), VGPR r -> M row r + 8*(lane>=16)
  const int cn    = lane & 15;
  const int rbase = (lane >> 4) << 3;
  float* crow = C + (size_t)(m0 + wtm + rbase) * N + n0 + wtn + cn;
#pragma unroll
  for (int mi = 0; mi < 2; ++mi)
#pragma unroll
    for (int ni = 0; ni < 4; ++ni)
#pragma unroll
      for (int r = 0; r < 8; ++r)
        crow[(size_t)(mi * 16 + r) * N + ni * 16] = acc[mi][ni][r];
}

// ---------------------------------------------------------------------------
// Pack kernel: qkv fp32 [B,T,1536] -> RoPE + RMS-norm + f16 layouts:
//   Q [B,NH,T,HD] (pre-scaled by 1/sqrt(HD)), K [B,NKV,T,HD], Vt [B,NKV,HD,T]
// One wave per (b,t,slot); lane owns RoPE pair (dims 2*lane, 2*lane+1).
// ---------------------------------------------------------------------------
__global__ __launch_bounds__(256)
void pack_rope_rms(const float* __restrict__ qkv, _Float16* __restrict__ Q,
                   _Float16* __restrict__ Kb, _Float16* __restrict__ Vt) {
  const int tid   = threadIdx.x;
  const int lane  = tid & 31;
  const int wslot = blockIdx.x * 8 + (tid >> 5);   // [0, B*T*24)
  const int slot  = wslot % 24;                    // 16 q heads, 4 k, 4 v
  const int bt    = wslot / 24;
  const int t     = bt % TT;
  const int b     = bt / TT;

  const float* base = qkv + (size_t)bt * QKVN;
  const float* row;
  if (slot < 16)      row = base + slot * HD_;
  else if (slot < 20) row = base + CC + (slot - 16) * HD_;
  else                row = base + CC + NKV_ * HD_ + (slot - 20) * HD_;

  float x0 = row[2 * lane + 0];
  float x1 = row[2 * lane + 1];

  if (slot < 20) {
    // RoPE: pair i=lane, theta = t * base^(-2i/HD)
    float inv_freq = __expf(-(2.0f * (float)lane / (float)HD_) * logf(10000.0f));
    float th = (float)t * inv_freq;
    float c = cosf(th), s = sinf(th);
    float r0 = x0 * c - x1 * s;
    float r1 = x1 * c + x0 * s;
    // RMS over the 64 head dims (cross-lane reduce over 32 lanes)
    float ss = r0 * r0 + r1 * r1;
#pragma unroll
    for (int off = 16; off; off >>= 1) ss += __shfl_xor(ss, off, 32);
    float inv = rsqrtf(ss * (1.0f / (float)HD_) + 1e-6f);
    r0 *= inv; r1 *= inv;
    if (slot < 16) {
      r0 *= 0.125f; r1 *= 0.125f;   // fold 1/sqrt(HD) into q
      _Float16* q = Q + (((size_t)b * NH_ + slot) * TT + t) * HD_;
      q[2 * lane + 0] = (_Float16)r0;
      q[2 * lane + 1] = (_Float16)r1;
    } else {
      _Float16* k = Kb + (((size_t)b * NKV_ + (slot - 16)) * TT + t) * HD_;
      k[2 * lane + 0] = (_Float16)r0;
      k[2 * lane + 1] = (_Float16)r1;
    }
  } else {
    // V: transpose to [B,NKV,HD,T]
    _Float16* v = Vt + (((size_t)b * NKV_ + (slot - 20)) * HD_ + 2 * lane) * TT + t;
    v[0]  = (_Float16)x0;
    v[TT] = (_Float16)x1;
  }
}

// ---------------------------------------------------------------------------
// Causal flash attention, f16 WMMA, f32 accumulation.
// Block = 8 waves; wave owns 16 query rows; KV blocks of 32, per-wave loop
// bound (no block barriers). Only the final KV block is masked.
// Softmax split across both half-waves (lane L / L+16 own cols 0-15 / 16-31
// of row L). Output Y fp32 [B,T,NH,HD] == [B*T, C] row-major.
// ---------------------------------------------------------------------------
__global__ __launch_bounds__(256)
void attn_flash_wmma(const _Float16* __restrict__ Q, const _Float16* __restrict__ Kb,
                     const _Float16* __restrict__ Vt, float* __restrict__ Y) {
  struct Scr {
    float    S[16][36];    // row stride 144B -> 16B-aligned v4f chunks
    _Float16 P[16][40];    // row stride 80B  -> 16B-aligned v8h chunks
    float    fac[16];
  };
  __shared__ Scr scr[8];

  const int tid  = threadIdx.x;
  const int lane = tid & 31;
  const int wave = tid >> 5;
  Scr& sw = scr[wave];

  const int qb = blockIdx.x & 15;            // T/128 = 16 q-blocks
  const int h  = (blockIdx.x >> 4) & 15;     // NH = 16
  const int b  = blockIdx.x >> 8;
  const int g  = h >> 2;                     // NH/NKV = 4

  const int m0w  = qb * 128 + wave * 16;     // first absolute q row (token t)
  const int am   = lane & 15;
  const int koff = (lane >> 4) << 3;         // A-frag K chunk base (0/8)
  const int bk   = (lane >> 4) << 4;         // unused; B-frags use koff (0/8)
  const int rb   = (lane >> 4) << 3;         // C/D row base (0/8)
  const int cb16 = (lane >> 4) << 4;         // softmax col-half base (0/16)

  // Load Q fragments once (HD=64 -> 2 k-chunks of 32)
  const _Float16* qrow = Q + (((size_t)b * NH_ + h) * TT + (m0w + am)) * HD_;
  v16h qf[2];
#pragma unroll
  for (int kc = 0; kc < 2; ++kc) {
    v8h a0 = *(const v8h*)(qrow + kc * 32 + koff);
    v8h a1 = *(const v8h*)(qrow + kc * 32 + koff + 16);
    qf[kc] = CAT16(a0, a1);
  }

  const _Float16* Kbase = Kb + ((size_t)b * NKV_ + g) * (size_t)TT * HD_;
  const _Float16* Vbase = Vt + ((size_t)b * NKV_ + g) * (size_t)HD_ * TT;

  v8f  acc[4] = {};
  float m_run = -3.0e38f;   // identical in lane L and L+16
  float l_run = 0.0f;

  auto kv_step = [&](int jb, bool masked) {
    // --- preload all K fragments, then 4 WMMA: S = Q * K^T (16x32) ---
    v16h bf[2][2];
#pragma unroll
    for (int nf = 0; nf < 2; ++nf)
#pragma unroll
      for (int kc = 0; kc < 2; ++kc) {
        const _Float16* kp = Kbase + (size_t)(jb + nf * 16 + am) * HD_ + kc * 32 + koff;
        v8h b0 = *(const v8h*)kp;
        v8h b1 = *(const v8h*)(kp + 16);
        bf[nf][kc] = CAT16(b0, b1);
      }
    v8f s[2] = {};
#pragma unroll
    for (int nf = 0; nf < 2; ++nf)
#pragma unroll
      for (int kc = 0; kc < 2; ++kc)
        s[nf] = WMMA_F16(qf[kc], bf[nf][kc], s[nf]);

    // dump S tile to per-wave LDS
#pragma unroll
    for (int nf = 0; nf < 2; ++nf)
#pragma unroll
      for (int r = 0; r < 8; ++r)
        sw.S[rb + r][nf * 16 + am] = s[nf][r];
    __builtin_amdgcn_wave_barrier();

    // --- preload V fragments now; latency hides behind softmax ---
    v16h vf[4];
#pragma unroll
    for (int nf = 0; nf < 4; ++nf) {
      const _Float16* vp = Vbase + (size_t)(nf * 16 + am) * TT + jb + koff;
      v8h b0 = *(const v8h*)vp;
      v8h b1 = *(const v8h*)(vp + 16);
      vf[nf] = CAT16(b0, b1);
    }

    // --- online softmax: row L split across lanes L (cols 0-15) and L+16 ---
    {
      v4f sv4[4];
#pragma unroll
      for (int jj = 0; jj < 4; ++jj)
        sv4[jj] = *(const v4f*)(&sw.S[am][cb16 + jj * 4]);
      float rmax = -3.0e38f;
      if (masked) {
        const int tv = (m0w + am) - jb - cb16;   // valid cols: j <= tv
#pragma unroll
        for (int j = 0; j < 16; ++j) {
          float v = (j <= tv) ? sv4[j >> 2][j & 3] : -3.0e38f;
          sv4[j >> 2][j & 3] = v;
          rmax = fmaxf(rmax, v);
        }
      } else {
#pragma unroll
        for (int j = 0; j < 16; ++j) rmax = fmaxf(rmax, sv4[j >> 2][j & 3]);
      }
      rmax = fmaxf(rmax, __shfl_xor(rmax, 16, 32));
      float mnew = fmaxf(m_run, rmax);
      float f    = __expf(m_run - mnew);
      float ssum = 0.0f;
      v8h pv[2];
#pragma unroll
      for (int j = 0; j < 16; ++j) {
        float p = __expf(sv4[j >> 2][j & 3] - mnew);
        ssum += p;
        pv[j >> 3][j & 7] = (_Float16)p;
      }
      *(v8h*)(&sw.P[am][cb16 + 0]) = pv[0];
      *(v8h*)(&sw.P[am][cb16 + 8]) = pv[1];
      ssum += __shfl_xor(ssum, 16, 32);
      l_run = l_run * f + ssum;
      m_run = mnew;
      if (lane < 16) sw.fac[am] = f;
    }
    __builtin_amdgcn_wave_barrier();

    // rescale running accumulator by per-row factors
#pragma unroll
    for (int r = 0; r < 8; ++r) {
      float f = sw.fac[rb + r];
#pragma unroll
      for (int nf = 0; nf < 4; ++nf) acc[nf][r] *= f;
    }

    // --- O += P * V (4 WMMA) ---
    v8h p0 = *(const v8h*)(&sw.P[am][koff]);
    v8h p1 = *(const v8h*)(&sw.P[am][koff + 16]);
    v16h pf = CAT16(p0, p1);
#pragma unroll
    for (int nf = 0; nf < 4; ++nf)
      acc[nf] = WMMA_F16(pf, vf[nf], acc[nf]);
  };

  // full (unmasked) KV blocks, then exactly one masked diagonal block
  const int nblk = ((m0w + 15) >> 5) + 1;
  for (int jblk = 0; jblk < nblk - 1; ++jblk) kv_step(jblk * 32, false);
  kv_step((nblk - 1) * 32, true);

  // final normalization by row sums, write Y [B,T,NH,HD]
  if (lane < 16) sw.fac[am] = 1.0f / l_run;
  __builtin_amdgcn_wave_barrier();
#pragma unroll
  for (int r = 0; r < 8; ++r) {
    float f = sw.fac[rb + r];
    const int t = m0w + rb + r;
#pragma unroll
    for (int nf = 0; nf < 4; ++nf) {
      int d = nf * 16 + am;
      Y[(((size_t)b * TT + t) * NH_ + h) * HD_ + d] = acc[nf][r] * f;
    }
  }
}

// ---------------------------------------------------------------------------
// Launch
// ---------------------------------------------------------------------------
extern "C" void kernel_launch(void* const* d_in, const int* in_sizes, int n_in,
                              void* d_out, int out_size, void* d_ws, size_t ws_size,
                              hipStream_t stream) {
  (void)in_sizes; (void)n_in; (void)out_size; (void)ws_size;
  const float* x      = (const float*)d_in[0];
  const float* w_attn = (const float*)d_in[1];
  const float* w_proj = (const float*)d_in[2];
  float* out = (float*)d_out;

  // Workspace layout (all offsets 256B-aligned)
  char* wsb = (char*)d_ws;
  const size_t SZ_QKV = (size_t)BB * TT * QKVN * sizeof(float);          // 25,165,824
  const size_t SZ_Q   = (size_t)BB * NH_  * TT * HD_ * sizeof(_Float16); //  8,388,608
  const size_t SZ_K   = (size_t)BB * NKV_ * TT * HD_ * sizeof(_Float16); //  2,097,152
  const size_t SZ_V   = SZ_K;
  float*    qkv = (float*)wsb;
  _Float16* Qb  = (_Float16*)(wsb + SZ_QKV);
  _Float16* Kb  = (_Float16*)(wsb + SZ_QKV + SZ_Q);
  _Float16* Vt  = (_Float16*)(wsb + SZ_QKV + SZ_Q + SZ_K);
  float*    Yb  = (float*)(wsb + SZ_QKV + SZ_Q + SZ_K + SZ_V);

  // 1) qkv = x @ w_attn   [4096,1024] x [1024,1536]
  gemm_wmma_f16<QKVN, CC><<<dim3(QKVN / 128, (BB * TT) / 128), 256, 0, stream>>>(
      x, w_attn, qkv);

  // 2) RoPE + RMS-norm + f16 packing (B*T*24 waves / 8 per block)
  pack_rope_rms<<<(BB * TT * 24) / 8, 256, 0, stream>>>(qkv, Qb, Kb, Vt);

  // 3) causal flash attention: B*NH*(T/128) = 512 blocks
  attn_flash_wmma<<<BB * NH_ * (TT / 128), 256, 0, stream>>>(Qb, Kb, Vt, Yb);

  // 4) out = Y @ w_proj   [4096,1024] x [1024,1024]
  gemm_wmma_f16<CC, CC><<<dim3(CC / 128, (BB * TT) / 128), 256, 0, stream>>>(
      Yb, w_proj, out);
}
